// Generator_37890201485765
// MI455X (gfx1250) — matrix operand order repaired and checked by reading, not compiled, not run on hardware
//
#include <hip/hip_runtime.h>

typedef _Float16 f16;
typedef __attribute__((ext_vector_type(16))) _Float16 v16h;
typedef __attribute__((ext_vector_type(8)))  _Float16 v8h;
typedef __attribute__((ext_vector_type(8)))  float    v8f;

#define LRELU_SLOPE 0.2f
#define EPS_DEMOD   1e-8f

// ---------------------------------------------------------------------------
// Small dense kernels
// ---------------------------------------------------------------------------

// sv[b*OD+o] = styles[b, blk, :] . Wl[o, :] + bl[o]    (latent = 512)
__global__ void style_linear_k(const float* __restrict__ styles,
                               const float* __restrict__ Wl,
                               const float* __restrict__ bl,
                               float* __restrict__ sv,
                               int blk, int B, int OD) {
  int i = blockIdx.x * blockDim.x + threadIdx.x;
  if (i >= B * OD) return;
  int o = i % OD, b = i / OD;
  const float* s = styles + ((size_t)b * 4 + blk) * 512;
  const float* w = Wl + (size_t)o * 512;
  float acc = bl[o];
  for (int k = 0; k < 512; ++k) acc = fmaf(s[k], w[k], acc);
  sv[i] = acc;
}

// dm[b*OC+o] = rsqrt( sum_{c,k} (W[o,c,k] * (sv[b,c]+1))^2 + eps )
__global__ void demod_k(const float* __restrict__ Wc, const float* __restrict__ sv,
                        float* __restrict__ dm, int B, int OC, int C, int K) {
  int i = blockIdx.x * blockDim.x + threadIdx.x;
  if (i >= B * OC) return;
  int o = i % OC, b = i / OC;
  const float* sb = sv + (size_t)b * C;
  int kk = K * K;
  float acc = 0.f;
  for (int c = 0; c < C; ++c) {
    float s = sb[c] + 1.f;
    const float* wp = Wc + ((size_t)o * C + c) * kk;
    for (int k = 0; k < kk; ++k) { float w = wp[k] * s; acc = fmaf(w, w, acc); }
  }
  dm[i] = rsqrtf(acc + EPS_DEMOD);
}

// NCHW f32 -> padded NHWC f16, scaled by (sv[b,c]+1); halo (1px) = 0.
__global__ void modulate_pad_k(const float* __restrict__ x, const float* __restrict__ sv,
                               f16* __restrict__ out, int B, int C, int H, int W) {
  size_t i = (size_t)blockIdx.x * blockDim.x + threadIdx.x;
  size_t total = (size_t)B * (H + 2) * (W + 2) * C;
  if (i >= total) return;
  int c = (int)(i % C); size_t t = i / C;
  int px = (int)(t % (W + 2)); t /= (W + 2);
  int py = (int)(t % (H + 2)); int b = (int)(t / (H + 2));
  float v = 0.f;
  int yy = py - 1, xx = px - 1;
  if (yy >= 0 && yy < H && xx >= 0 && xx < W) {
    float s = sv ? (sv[(size_t)b * C + c] + 1.f) : 1.f;
    v = x[(((size_t)b * C + c) * H + yy) * W + xx] * s;
  }
  out[i] = (f16)v;
}

// OIHW f32 weights -> f16 [tap][ocPad][ic] (rows >= OC zero-filled)
__global__ void wxform_k(const float* __restrict__ Wc, f16* __restrict__ Wk,
                         int OC, int OCp, int C, int K) {
  size_t i = (size_t)blockIdx.x * blockDim.x + threadIdx.x;
  size_t total = (size_t)K * K * OCp * C;
  if (i >= total) return;
  int c = (int)(i % C); size_t t = i / C;
  int m = (int)(t % OCp); int kp = (int)(t / OCp);
  f16 v = (f16)0.f;
  if (m < OC) v = (f16)Wc[(((size_t)m * C + c) * K + kp / K) * K + kp % K];
  Wk[i] = v;
}

// bilinear 2x upsample, NCHW f32, half-pixel centers, edge clamp
__global__ void upsample2x_k(const float* __restrict__ in, float* __restrict__ out,
                             int B, int C, int H, int W) {
  int OH = 2 * H, OW = 2 * W;
  size_t i = (size_t)blockIdx.x * blockDim.x + threadIdx.x;
  size_t total = (size_t)B * C * OH * OW;
  if (i >= total) return;
  int ox = (int)(i % OW); size_t t = i / OW;
  int oy = (int)(t % OH); t /= OH;
  int c = (int)(t % C); int b = (int)(t / C);
  float sy = oy * 0.5f - 0.25f, sx = ox * 0.5f - 0.25f;
  int y0 = (int)floorf(sy), x0 = (int)floorf(sx);
  float fy = sy - (float)y0, fx = sx - (float)x0;
  int y1 = y0 + 1, x1 = x0 + 1;
  y0 = min(max(y0, 0), H - 1); y1 = min(max(y1, 0), H - 1);
  x0 = min(max(x0, 0), W - 1); x1 = min(max(x1, 0), W - 1);
  const float* p = in + (((size_t)b * C + c) * H) * W;
  float v00 = p[(size_t)y0 * W + x0], v01 = p[(size_t)y0 * W + x1];
  float v10 = p[(size_t)y1 * W + x0], v11 = p[(size_t)y1 * W + x1];
  float top = v00 + fx * (v01 - v00);
  float bot = v10 + fx * (v11 - v10);
  out[i] = top + fy * (bot - top);
}

// depthwise 3x3 blur [1,2,1]x[1,2,1]/16, zero padding, NCHW f32
__global__ void blur3_k(const float* __restrict__ in, float* __restrict__ out,
                        int B, int C, int H, int W) {
  size_t i = (size_t)blockIdx.x * blockDim.x + threadIdx.x;
  size_t total = (size_t)B * C * H * W;
  if (i >= total) return;
  int x = (int)(i % W); size_t t = i / W;
  int y = (int)(t % H); t /= H;
  const float* p = in + t * (size_t)H * W;
  float acc = 0.f;
  const float wgt[3] = {1.f, 2.f, 1.f};
  for (int dy = -1; dy <= 1; ++dy) {
    int yy = y + dy;
    if (yy < 0 || yy >= H) continue;
    float rowv = 0.f;
    for (int dx = -1; dx <= 1; ++dx) {
      int xx = x + dx;
      if (xx < 0 || xx >= W) continue;
      rowv += wgt[dx + 1] * p[(size_t)yy * W + xx];
    }
    acc += wgt[dy + 1] * rowv;
  }
  out[i] = acc * (1.f / 16.f);
}

// ---------------------------------------------------------------------------
// WMMA implicit-GEMM convolution (templated, division-free).
//   Min : padded NHWC f16 modulated input, (B, H+2, W+2, C)
//   Wk  : f16 weights [KS*KS][OCp][C]
//   out : NCHW f32 (B, OC, H, W)
// Grid: x = ceil(tilesX*(H/4)/8) (8 waves/WG), y = OCp/(16*MT), z = B.
// Each wave: (MT*16) oc x 16 px x 4 rows. The 4 B fragments are reused
// across MT A fragments -> MT*4 WMMAs per K-chunk.
// MODE: 0=+bias, 1=demod+noise+lrelu, 2=rgb(+prev).
// ---------------------------------------------------------------------------
template <int KS, int MODE, int MT>
__global__ __launch_bounds__(256)
void conv_wmma_k(const f16* __restrict__ Min, const f16* __restrict__ Wk,
                 float* __restrict__ out,
                 const float* __restrict__ dm, const float* __restrict__ inoise,
                 const float* __restrict__ nw, const float* __restrict__ nb,
                 const float* __restrict__ bias, const float* __restrict__ prev,
                 int C, int OC, int OCp, int H, int W,
                 int xtShift, int nTiles) {
  int wave = threadIdx.x >> 5;
  int lane = threadIdx.x & 31;
  int tid  = blockIdx.x * 8 + wave;
  if (tid >= nTiles) return;                 // wave-uniform exit

  int tilesXm1 = (1 << xtShift) - 1;
  int xt = tid & tilesXm1;                   // W/16 is a power of two
  int yg = tid >> xtShift;
  int mbase = blockIdx.y * (16 * MT);
  int b  = blockIdx.z;
  int y0 = yg << 2;

  constexpr int SH = 1 - ((KS - 1) >> 1);    // padded-coord offset
  int Wp = W + 2;
  int khalf = lane >> 4;
  int nx = (xt << 4) + (lane & 15);          // output x this lane feeds

  const f16* a0 = Wk + (size_t)(mbase + (lane & 15)) * C + khalf * 16;
  size_t aTap = (size_t)OCp * C;
  size_t rowStride = (size_t)Wp * C;
  const f16* bbase = Min + (((size_t)b * (H + 2) + (y0 + SH)) * Wp + (nx + SH)) * C
                   + khalf * 16;

  v8f acc[MT][4] = {};
#pragma unroll
  for (int kp = 0; kp < KS * KS; ++kp) {
    const int kh = kp / KS, kw = kp % KS;
    const f16* arow = a0 + (size_t)kp * aTap;
    const f16* brow = bbase + (size_t)kh * rowStride + (size_t)kw * C;
#pragma unroll 4
    for (int c = 0; c < C; c += 32) {
      __builtin_prefetch((const void*)(brow + c + 256), 0, 1);  // global_prefetch
      v16h av[MT];
#pragma unroll
      for (int mi = 0; mi < MT; ++mi) {
        union { v16h v; v8h h[2]; } ua;       // K = 16*khalf + j, contiguous
        ua.h[0] = *(const v8h*)(arow + (size_t)mi * 16 * C + c);
        ua.h[1] = *(const v8h*)(arow + (size_t)mi * 16 * C + c + 8);
        av[mi] = ua.v;
      }
      v16h bv[4];
#pragma unroll
      for (int ri = 0; ri < 4; ++ri)          // K = 16*khalf + j, contiguous
        bv[ri] = *(const v16h*)(brow + (size_t)ri * rowStride + c);
#pragma unroll
      for (int mi = 0; mi < MT; ++mi)
#pragma unroll
        for (int ri = 0; ri < 4; ++ri)
          acc[mi][ri] = __builtin_amdgcn_wmma_f32_16x16x32_f16(
              false, av[mi], false, bv[ri], (short)0, acc[mi][ri], false, false);
    }
  }

  // epilogue: lane holds N = lane&15 (x = nx); rows m = mbase + mi*16 + r + 8*khalf
#pragma unroll
  for (int mi = 0; mi < MT; ++mi) {
#pragma unroll
    for (int rr = 0; rr < 4; ++rr) {
      int y = y0 + rr;
#pragma unroll
      for (int r = 0; r < 8; ++r) {
        int mo = mbase + mi * 16 + r + 8 * khalf;
        if (mo >= OC) continue;
        float v = acc[mi][rr][r];
        if (MODE == 1) {
          v *= dm[(size_t)b * OC + mo];
          // reference transposes spatial dims of the noise: n[b, x, y]
          float nz = inoise[((size_t)b * 256 + nx) * 256 + y];
          v += nz * nw[mo] + nb[mo];
          v = (v >= 0.f) ? v : LRELU_SLOPE * v;
        } else if (MODE == 0) {
          v += bias[mo];
        } else {  // MODE 2: to-RGB accumulate
          if (prev) v += prev[(((size_t)b * OC + mo) * H + y) * W + nx];
        }
        out[(((size_t)b * OC + mo) * H + y) * W + nx] = v;
      }
    }
  }
}

// ---------------------------------------------------------------------------
// Host orchestration
// ---------------------------------------------------------------------------
static inline int gdiv(size_t n) { return (int)((n + 255) / 256); }
static inline int ilog2(int v) { int s = 0; while ((1 << s) < v) ++s; return s; }

extern "C" void kernel_launch(void* const* d_in, const int* in_sizes, int n_in,
                              void* d_out, int out_size, void* d_ws, size_t ws_size,
                              hipStream_t stream) {
  (void)in_sizes; (void)n_in; (void)out_size;
  const int B = 4;
  const float* styles  = (const float*)d_in[0];   // (4,4,512)
  const float* inoise  = (const float*)d_in[1];   // (4,256,256,1)
  const float* s_input = (const float*)d_in[2];   // (4,512,32,32)
  const float* init_w  = (const float*)d_in[3];   // (512,512,3,3)
  const float* init_b  = (const float*)d_in[4];   // (512,)

  // ---- workspace carve ----
  const size_t actElems = (size_t)4 * 64 * 256 * 256;        // 16.78M f32
  const size_t mElems   = (size_t)4 * 258 * 258 * 64;        // 17.04M f16
  const size_t wkElems  = (size_t)9 * 512 * 512;             // 2.36M  f16
  const size_t rgbElems = (size_t)4 * 3 * 256 * 256;
  size_t need = actElems * 2 * 4 + mElems * 2 + wkElems * 2 +
                (2048 + 1024 + rgbElems * 2) * 4;
  if (ws_size < need) return;

  char* ws = (char*)d_ws;
  float* X    = (float*)ws;  ws += actElems * 4;
  float* Y    = (float*)ws;  ws += actElems * 4;
  f16*   M    = (f16*)ws;    ws += mElems * 2;
  f16*   Wk   = (f16*)ws;    ws += wkElems * 2;
  float* sv   = (float*)ws;  ws += 2048 * 4;
  float* dm   = (float*)ws;  ws += 1024 * 4;
  float* rgbN = (float*)ws;  ws += rgbElems * 4;
  float* rgbP = (float*)ws;  ws += rgbElems * 4;

  auto convGrid = [&](int OCp, int MT, int H, int W, int& nTiles, int& xtShift) {
    int tilesX = W >> 4;
    xtShift = ilog2(tilesX);
    nTiles = tilesX * (H >> 2);
    return dim3((unsigned)((nTiles + 7) / 8), (unsigned)(OCp / (16 * MT)),
                (unsigned)B);
  };

  // ---- init conv: x = conv3x3(s_input, init_w) + init_b ----
  {
    int C = 512, OC = 512, H = 32, W = 32;
    wxform_k<<<gdiv((size_t)9 * OC * C), 256, 0, stream>>>(init_w, Wk, OC, OC, C, 3);
    modulate_pad_k<<<gdiv((size_t)B * (H + 2) * (W + 2) * C), 256, 0, stream>>>(
        s_input, nullptr, M, B, C, H, W);
    int nT, xs; dim3 g = convGrid(OC, 2, H, W, nT, xs);
    conv_wmma_k<3, 0, 2><<<g, 256, 0, stream>>>(M, Wk, X, nullptr, inoise,
                                                nullptr, nullptr, init_b,
                                                nullptr, C, OC, OC, H, W, xs, nT);
  }

  const int ICt[4] = {512, 256, 128, 64};
  const int OCt[4] = {256, 128, 64, 32};
  float* cur = X;
  float* oth = Y;

  for (int i = 0; i < 4; ++i) {
    const float* const* p = (const float* const*)(d_in + 5 + 13 * i);
    const float *s1w = p[0], *s1b = p[1], *n1w = p[2], *n1b = p[3], *c1w = p[4];
    const float *s2w = p[5], *s2b = p[6], *n2w = p[7], *n2b = p[8], *c2w = p[9];
    const float *rsw = p[10], *rsb = p[11], *rcw = p[12];
    int ic = ICt[i], oc = OCt[i];
    int H = 32 << i, W = 32 << i;

    if (i > 0) {  // upsample x
      upsample2x_k<<<gdiv((size_t)B * ic * H * W), 256, 0, stream>>>(
          cur, oth, B, ic, H / 2, W / 2);
      float* tmp = cur; cur = oth; oth = tmp;
    }

    // conv1: ic -> oc, demod + noise1 + lrelu
    style_linear_k<<<gdiv((size_t)B * ic), 256, 0, stream>>>(styles, s1w, s1b, sv, i, B, ic);
    demod_k<<<gdiv((size_t)B * oc), 256, 0, stream>>>(c1w, sv, dm, B, oc, ic, 3);
    modulate_pad_k<<<gdiv((size_t)B * (H + 2) * (W + 2) * ic), 256, 0, stream>>>(
        cur, sv, M, B, ic, H, W);
    wxform_k<<<gdiv((size_t)9 * oc * ic), 256, 0, stream>>>(c1w, Wk, oc, oc, ic, 3);
    {
      int nT, xs; dim3 g = convGrid(oc, 2, H, W, nT, xs);
      conv_wmma_k<3, 1, 2><<<g, 256, 0, stream>>>(M, Wk, oth, dm, inoise, n1w,
                                                  n1b, nullptr, nullptr, ic, oc,
                                                  oc, H, W, xs, nT);
    }
    { float* tmp = cur; cur = oth; oth = tmp; }   // cur = x1

    // conv2: oc -> oc, demod + noise2 + lrelu
    style_linear_k<<<gdiv((size_t)B * oc), 256, 0, stream>>>(styles, s2w, s2b, sv, i, B, oc);
    demod_k<<<gdiv((size_t)B * oc), 256, 0, stream>>>(c2w, sv, dm, B, oc, oc, 3);
    modulate_pad_k<<<gdiv((size_t)B * (H + 2) * (W + 2) * oc), 256, 0, stream>>>(
        cur, sv, M, B, oc, H, W);
    wxform_k<<<gdiv((size_t)9 * oc * oc), 256, 0, stream>>>(c2w, Wk, oc, oc, oc, 3);
    {
      int nT, xs; dim3 g = convGrid(oc, 2, H, W, nT, xs);
      conv_wmma_k<3, 1, 2><<<g, 256, 0, stream>>>(M, Wk, oth, dm, inoise, n2w,
                                                  n2b, nullptr, nullptr, oc, oc,
                                                  oc, H, W, xs, nT);
    }
    { float* tmp = cur; cur = oth; oth = tmp; }   // cur = x2

    // to-RGB: oc -> 3 (padded to 16 rows), 1x1, modulated, no demod, + prev
    style_linear_k<<<gdiv((size_t)B * oc), 256, 0, stream>>>(styles, rsw, rsb, sv, i, B, oc);
    modulate_pad_k<<<gdiv((size_t)B * (H + 2) * (W + 2) * oc), 256, 0, stream>>>(
        cur, sv, M, B, oc, H, W);
    wxform_k<<<gdiv((size_t)16 * oc), 256, 0, stream>>>(rcw, Wk, 3, 16, oc, 1);
    float* rout = (i == 3) ? (float*)d_out : rgbN;
    {
      int nT, xs; dim3 g = convGrid(16, 1, H, W, nT, xs);
      conv_wmma_k<1, 2, 1><<<g, 256, 0, stream>>>(M, Wk, rout, nullptr, inoise,
                                                  nullptr, nullptr, nullptr,
                                                  (i > 0) ? rgbP : nullptr,
                                                  oc, 3, 16, H, W, xs, nT);
    }

    if (i < 3) {  // rgb = blur(upsample2x(rgb)); free act buffer as temp
      upsample2x_k<<<gdiv((size_t)B * 3 * 4 * H * W), 256, 0, stream>>>(
          rgbN, oth, B, 3, H, W);
      blur3_k<<<gdiv((size_t)B * 3 * 4 * H * W), 256, 0, stream>>>(
          oth, rgbP, B, 3, 2 * H, 2 * W);
    }
  }
}
